// DEQ_Configurable_2611340116040
// MI455X (gfx1250) — matrix operand order repaired and checked by reading, not compiled, not run on hardware
//
#include <hip/hip_runtime.h>
#include <math.h>
#include <stdint.h>

// ---------------------------------------------------------------------------
// DEQ forward on MI455X (gfx1250): bf16 WMMA GEMMs + async-to-LDS staging
// (GLOBAL_LOAD_ASYNC_TO_LDS_B128, ASYNCcnt-tracked, LDS double buffered)
// + fused LN/tanh iteration.
// ---------------------------------------------------------------------------

#ifndef USE_ASYNC_COPY
#define USE_ASYNC_COPY 1
#endif

typedef __bf16 v16bf __attribute__((ext_vector_type(16)));
typedef float  v8f   __attribute__((ext_vector_type(8)));

enum { Bsz = 4096, IN_DIM = 1024, HID = 2048, OUT_DIM = 1024 };
enum { SOLVER_ITERS = 24, PHANTOM_K = 5 };
#define TOTAL_ITERS (SOLVER_ITERS + PHANTOM_K)

typedef unsigned short ushort_t;

static __device__ __forceinline__ ushort_t f32_to_bf16(float f) {
    unsigned int u = __float_as_uint(f);
    unsigned int rounding = 0x7FFFu + ((u >> 16) & 1u);
    return (ushort_t)((u + rounding) >> 16);
}

// --- CDNA5 async copy: 16B global -> LDS per lane, tracked by ASYNCcnt -----
// Low 32 bits of a generic LDS pointer are the DS address (ISA 10.2 aperture
// table).
static __device__ __forceinline__ void async_cp16(void* lds_ptr, const void* gaddr) {
#if USE_ASYNC_COPY
    const uint32_t lds_addr = (uint32_t)(uintptr_t)lds_ptr;
    asm volatile("global_load_async_to_lds_b128 %0, %1, off"
                 :: "v"(lds_addr), "v"(gaddr)
                 : "memory");
#else
    *(uint4*)lds_ptr = *(const uint4*)gaddr;
#endif
}

#if __has_builtin(__builtin_amdgcn_s_wait_asynccnt)
#define WAIT_ASYNC(n) __builtin_amdgcn_s_wait_asynccnt(n)
#else
#define WAIT_ASYNC(n) asm volatile("s_wait_asynccnt %0" :: "i"(n) : "memory")
#endif

// ------------------------- f32 -> bf16 conversion --------------------------
__global__ __launch_bounds__(256) void cvt_f32_bf16(const float* __restrict__ src,
                                                    ushort_t* __restrict__ dst, int n) {
    for (int i = blockIdx.x * blockDim.x + threadIdx.x; i < n; i += gridDim.x * blockDim.x)
        dst[i] = f32_to_bf16(src[i]);
}

// ------------------------------ WMMA GEMM ----------------------------------
// C[M,N] = A[M,K](bf16) @ W[N,K](bf16)^T  (+ bias[n]) (+ addm[m,n]),
// output either f32 or bf16.  Block tile 128x128x32, 256 threads = 8 waves,
// LDS double buffered with async global->LDS copies.
#define BM 128
#define BN 128
#define BK 32
#define LDSS 40   // LDS row stride in bf16 elems (32 + 8 pad, keeps 16B align)

union FragU {
    v16bf v;
    uint4 q[2];
};

template <bool OUT_BF16, bool HAS_BIAS, bool HAS_ADD>
__global__ __launch_bounds__(256) void gemm_bf16(
    const ushort_t* __restrict__ A, int lda,
    const ushort_t* __restrict__ W, int ldw,
    const float* __restrict__ bias,
    const float* __restrict__ addm,
    float* __restrict__ outF, ushort_t* __restrict__ outB,
    int N, int K) {
    __shared__ ushort_t sA[2][BM * LDSS];
    __shared__ ushort_t sW[2][BN * LDSS];

    const int tid   = threadIdx.x;
    const int lane  = tid & 31;
    const int wv    = tid >> 5;
    const int waveM = wv & 1;   // 2 waves over M (64 rows each)
    const int waveN = wv >> 1;  // 4 waves over N (32 cols each)
    const int rowBase = blockIdx.y * BM;
    const int colBase = blockIdx.x * BN;

    v8f acc[4][2];
#pragma unroll
    for (int i = 0; i < 4; ++i)
#pragma unroll
        for (int j = 0; j < 2; ++j) acc[i][j] = (v8f){0.f, 0.f, 0.f, 0.f, 0.f, 0.f, 0.f, 0.f};

    // Stage one 128x32 A tile + 128x32 W tile into LDS buffer `buf`.
    // 512 chunks of 8 bf16 per tile; 2 chunks/thread -> 4 async ops/thread.
    auto stage = [&](int buf, int k0) {
#pragma unroll
        for (int c = 0; c < 2; ++c) {
            const int ch = tid + c * 256;
            const int r  = ch >> 2;          // 0..127
            const int kc = (ch & 3) * 8;     // 0,8,16,24
            async_cp16(&sA[buf][r * LDSS + kc],
                       A + (size_t)(rowBase + r) * lda + k0 + kc);
            async_cp16(&sW[buf][r * LDSS + kc],
                       W + (size_t)(colBase + r) * ldw + k0 + kc);
        }
    };

    // One WMMA K=32 step from LDS buffer `buf`.
    auto compute = [&](int buf) {
        // B fragments: lane n<16 -> col n, K[0..15]; lane n+16 -> K[16..31]
        FragU fb[2];
#pragma unroll
        for (int ni = 0; ni < 2; ++ni) {
            const int col = waveN * 32 + ni * 16 + (lane & 15);
            const int kb  = (lane < 16) ? 0 : 16;
            fb[ni].q[0] = *(const uint4*)(&sW[buf][col * LDSS + kb]);
            fb[ni].q[1] = *(const uint4*)(&sW[buf][col * LDSS + kb + 8]);
        }
        // A fragments: lane m<16 -> K[0..15]; lane m+16 -> K[16..31]
#pragma unroll
        for (int mi = 0; mi < 4; ++mi) {
            const int row = waveM * 64 + mi * 16 + (lane & 15);
            const int ka  = (lane < 16) ? 0 : 16;
            FragU fa;
            fa.q[0] = *(const uint4*)(&sA[buf][row * LDSS + ka]);
            fa.q[1] = *(const uint4*)(&sA[buf][row * LDSS + ka + 8]);
#pragma unroll
            for (int ni = 0; ni < 2; ++ni) {
                acc[mi][ni] = __builtin_amdgcn_wmma_f32_16x16x32_bf16(
                    false, fa.v, false, fb[ni].v, (short)0, acc[mi][ni], false, false);
            }
        }
    };

#if USE_ASYNC_COPY
    // Double-buffered pipeline: issue tile k+1 async while computing tile k.
    stage(0, 0);
    int cur = 0;
    for (int k0 = 0; k0 < K; k0 += BK) {
        if (k0 + BK < K) {
            stage(cur ^ 1, k0 + BK);
            WAIT_ASYNC(4);   // in-order: the older group (4 ops/wave) retired
        } else {
            WAIT_ASYNC(0);
        }
        __syncthreads();     // all waves' copies for `cur` visible
        compute(cur);
        __syncthreads();     // done reading `cur` before it is restaged
        cur ^= 1;
    }
#else
    for (int k0 = 0; k0 < K; k0 += BK) {
        __syncthreads();
        stage(0, k0);
        __syncthreads();
        compute(0);
    }
#endif

    // Epilogue. C/D layout: VGPR r, lanes 0-15 => (M=r, N=lane); lanes 16-31 => (M=r+8, N=lane-16)
    const int nHalf  = lane & 15;
    const int rowOff = (lane & 16) ? 8 : 0;
#pragma unroll
    for (int mi = 0; mi < 4; ++mi) {
#pragma unroll
        for (int ni = 0; ni < 2; ++ni) {
            const int gcol = colBase + waveN * 32 + ni * 16 + nHalf;
            const float bv = HAS_BIAS ? bias[gcol] : 0.0f;
#pragma unroll
            for (int r = 0; r < 8; ++r) {
                const int grow = rowBase + waveM * 64 + mi * 16 + rowOff + r;
                const size_t idx = (size_t)grow * N + gcol;
                float v = acc[mi][ni][r] + bv;
                if (HAS_ADD) v += addm[idx];
                if (OUT_BF16) outB[idx] = f32_to_bf16(v);
                else          outF[idx] = v;
            }
        }
    }
}

// ----------------------- LayerNorm + tanh -> bf16 --------------------------
// One 256-thread block per row of h[Bsz, HID]; writes z in bf16.
__global__ __launch_bounds__(256) void ln_tanh_kernel(const float* __restrict__ h,
                                                      const float* __restrict__ g,
                                                      const float* __restrict__ b,
                                                      ushort_t* __restrict__ z) {
    __shared__ float redS[8];
    __shared__ float redQ[8];
    const int row  = blockIdx.x;
    const int tid  = threadIdx.x;
    const int lane = tid & 31;
    const int wv   = tid >> 5;
    const float* hr = h + (size_t)row * HID;
    ushort_t*    zr = z + (size_t)row * HID;

    float v[8];
    float s = 0.f, q = 0.f;
#pragma unroll
    for (int i = 0; i < 8; ++i) {
        const int c = tid + i * 256;
        const float x = hr[c];
        v[i] = x;
        s += x;
        q += x * x;
    }
    // wave32 reduction
#pragma unroll
    for (int off = 16; off > 0; off >>= 1) {
        s += __shfl_xor(s, off, 32);
        q += __shfl_xor(q, off, 32);
    }
    if (lane == 0) { redS[wv] = s; redQ[wv] = q; }
    __syncthreads();
    float st = 0.f, qt = 0.f;
#pragma unroll
    for (int i = 0; i < 8; ++i) { st += redS[i]; qt += redQ[i]; }

    const float inv  = 1.0f / (float)HID;
    const float mu   = st * inv;
    const float var  = qt * inv - mu * mu;
    const float rstd = rsqrtf(var + 1e-5f);
#pragma unroll
    for (int i = 0; i < 8; ++i) {
        const int c = tid + i * 256;
        const float t = tanhf((v[i] - mu) * rstd * g[c] + b[c]);
        zr[c] = f32_to_bf16(t);
    }
}

// ------------------------------- driver ------------------------------------
extern "C" void kernel_launch(void* const* d_in, const int* in_sizes, int n_in,
                              void* d_out, int out_size, void* d_ws, size_t ws_size,
                              hipStream_t stream) {
    (void)in_sizes; (void)n_in; (void)out_size; (void)ws_size;
    const float* x_f   = (const float*)d_in[0];
    const float* piw_f = (const float*)d_in[1];
    const float* pib   = (const float*)d_in[2];
    const float* wzw_f = (const float*)d_in[3];
    const float* wz_b  = (const float*)d_in[4];
    const float* wxw_f = (const float*)d_in[5];
    const float* ln_g  = (const float*)d_in[6];
    const float* ln_b  = (const float*)d_in[7];
    const float* hdw_f = (const float*)d_in[8];
    const float* hd_b  = (const float*)d_in[9];
    float* out = (float*)d_out;

    // Workspace carve-up (256B aligned)
    char* ws = (char*)d_ws;
    size_t off = 0;
    auto carve = [&](size_t bytes) -> void* {
        void* p = ws + off;
        off = (off + bytes + 255) & ~(size_t)255;
        return p;
    };
    ushort_t* x_b   = (ushort_t*)carve((size_t)Bsz * IN_DIM * 2);
    ushort_t* piw_b = (ushort_t*)carve((size_t)HID * IN_DIM * 2);
    ushort_t* wxw_b = (ushort_t*)carve((size_t)HID * HID * 2);
    ushort_t* wzw_b = (ushort_t*)carve((size_t)HID * HID * 2);
    ushort_t* hdw_b = (ushort_t*)carve((size_t)OUT_DIM * HID * 2);
    ushort_t* xpz_b = (ushort_t*)carve((size_t)Bsz * HID * 2);  // xp, then reused as z
    float*    xc    = (float*)carve((size_t)Bsz * HID * 4);
    float*    hbuf  = (float*)carve((size_t)Bsz * HID * 4);

    // 1) Convert inputs/weights to bf16
    cvt_f32_bf16<<<2048, 256, 0, stream>>>(x_f, x_b, Bsz * IN_DIM);
    cvt_f32_bf16<<<2048, 256, 0, stream>>>(piw_f, piw_b, HID * IN_DIM);
    cvt_f32_bf16<<<2048, 256, 0, stream>>>(wxw_f, wxw_b, HID * HID);
    cvt_f32_bf16<<<2048, 256, 0, stream>>>(wzw_f, wzw_b, HID * HID);
    cvt_f32_bf16<<<2048, 256, 0, stream>>>(hdw_f, hdw_b, OUT_DIM * HID);

    // 2) xp = x @ proj_in_w^T + proj_in_b   -> bf16  [4096, 2048]
    gemm_bf16<true, true, false><<<dim3(HID / BN, Bsz / BM), 256, 0, stream>>>(
        x_b, IN_DIM, piw_b, IN_DIM, pib, nullptr, nullptr, xpz_b, HID, IN_DIM);

    // 3) xc = xp @ wx_w^T                   -> f32   [4096, 2048]
    gemm_bf16<false, false, false><<<dim3(HID / BN, Bsz / BM), 256, 0, stream>>>(
        xpz_b, HID, wxw_b, HID, nullptr, nullptr, xc, nullptr, HID, HID);

    // 4) z0 = 0  (reuse xp buffer as z, bf16 zero == 0x0000)
    hipMemsetAsync(xpz_b, 0, (size_t)Bsz * HID * 2, stream);

    // 5) 29x:  h = z @ wz_w^T + wz_b + xc ;  z = tanh(LN(h))
    for (int it = 0; it < TOTAL_ITERS; ++it) {
        gemm_bf16<false, true, true><<<dim3(HID / BN, Bsz / BM), 256, 0, stream>>>(
            xpz_b, HID, wzw_b, HID, wz_b, xc, hbuf, nullptr, HID, HID);
        ln_tanh_kernel<<<Bsz, 256, 0, stream>>>(hbuf, ln_g, ln_b, xpz_b);
    }

    // 6) out = z @ head_w^T + head_b        -> f32   [4096, 1024]
    gemm_bf16<false, true, false><<<dim3(OUT_DIM / BN, Bsz / BM), 256, 0, stream>>>(
        xpz_b, HID, hdw_b, HID, hd_b, nullptr, out, nullptr, OUT_DIM, HID);
}
